// EthicalAttention_34772055228975
// MI455X (gfx1250) — matrix operand order repaired and checked by reading, not compiled, not run on hardware
//
#include <hip/hip_runtime.h>

typedef __attribute__((ext_vector_type(16))) _Float16 v16h;
typedef __attribute__((ext_vector_type(8)))  float    v8f;
typedef __attribute__((ext_vector_type(4)))  _Float16 v4h;

union Frag { v16h v; uint4 u[2]; };

constexpr int DM = 1024;   // d_model
constexpr int NH = 16;     // heads
constexpr int DK = 64;     // head dim
constexpr int BB = 2;      // batch
constexpr int SS = 2048;   // seq len

__device__ __forceinline__ v8f wmma_f16(const Frag& a, const Frag& b, v8f c) {
  return __builtin_amdgcn_wmma_f32_16x16x32_f16(false, a.v, false, b.v,
                                                (short)0, c, false, false);
}

// ---------------- f32 -> f16 conversion (vectorized x4) ----------------
__global__ void cvt_f32_f16_v4(const float4* __restrict__ src,
                               v4h* __restrict__ dst, int n4) {
  int i = blockIdx.x * blockDim.x + threadIdx.x;
  if (i < n4) {
    float4 v = src[i];
    v4h h;
    h.x = (_Float16)v.x; h.y = (_Float16)v.y;
    h.z = (_Float16)v.z; h.w = (_Float16)v.w;
    dst[i] = h;
  }
}

// ---------------- mask -> additive float bias (0 or -3e38) -------------
__global__ void mask_bias_kernel(const int* __restrict__ mask,
                                 float* __restrict__ kbias, int n) {
  int i = blockIdx.x * blockDim.x + threadIdx.x;
  if (i < n) kbias[i] = (mask[i] != 0) ? 0.f : -3.0e38f;
}

// ---------------- moral bias: mb[b,n] = mc[b,:] . Wmc[n,:] + bmc[n] ----
__global__ void moral_bias_kernel(const float* __restrict__ mc,
                                  const float* __restrict__ Wmc,
                                  const float* __restrict__ bmc,
                                  float* __restrict__ mb) {
  int i = blockIdx.x * blockDim.x + threadIdx.x;
  if (i >= BB * DM) return;
  int b = i / DM, n = i % DM;
  const float* mcr = mc + b * 64;
  const float* w   = Wmc + (size_t)n * 64;
  float s = bmc[n];
  for (int c = 0; c < 64; ++c) s += mcr[c] * w[c];
  mb[i] = s;
}

// ---------------- WMMA GEMM: out = A(MxK,f16) @ W(NxK,f16)^T + bias ----
// MODE 0: f32 row-major MxN        (final output projection)
// MODE 1: f16 -> (B,H,S,DK)        (Q, K for attention)
// MODE 2: f16 -> (B,H,DK,S)        (V transposed for attention)
template <int MODE>
__global__ void gemm_wmma(const _Float16* __restrict__ A,
                          const _Float16* __restrict__ W,
                          const float* __restrict__ bias,
                          float* __restrict__ outF,
                          _Float16* __restrict__ outH,
                          int M, int N, int K) {
  int lane = threadIdx.x & 31;
  int wid  = threadIdx.x >> 5;
  int l16  = lane & 15;
  int hi   = lane >> 4;                 // 0 or 1 (lane half)
  int m0 = blockIdx.x * 64 + (wid & 3) * 16;
  int n0 = blockIdx.y * 64 + (wid >> 2) * 32;

  v8f acc0 = {}; v8f acc1 = {};
  const _Float16* arow = A + (size_t)(m0 + l16) * K + hi * 16;
  const _Float16* br0  = W + (size_t)(n0 + l16) * K + hi * 16;
  const _Float16* br1  = W + (size_t)(n0 + 16 + l16) * K + hi * 16;

  for (int kt = 0; kt < K; kt += 32) {
    Frag a, b0, b1;
    a.u[0]  = *(const uint4*)(arow + kt);
    a.u[1]  = *(const uint4*)(arow + kt + 8);
    b0.u[0] = *(const uint4*)(br0 + kt);
    b0.u[1] = *(const uint4*)(br0 + kt + 8);
    b1.u[0] = *(const uint4*)(br1 + kt);
    b1.u[1] = *(const uint4*)(br1 + kt + 8);
    if (kt + 128 < K) {
      __builtin_prefetch(arow + kt + 128, 0, 0);
      __builtin_prefetch(br0 + kt + 128, 0, 0);
      __builtin_prefetch(br1 + kt + 128, 0, 0);
    }
    acc0 = wmma_f16(a, b0, acc0);
    acc1 = wmma_f16(a, b1, acc1);
  }

  for (int t = 0; t < 2; ++t) {
    v8f acc = t ? acc1 : acc0;
    int n = n0 + t * 16 + l16;
    float bn = bias[n];
    for (int r = 0; r < 8; ++r) {
      int m = m0 + r + 8 * hi;
      float val = acc[r] + bn;
      if (MODE == 0) {
        outF[(size_t)m * N + n] = val;
      } else {
        int b = m >> 11, s = m & (SS - 1);
        int h = n >> 6, d = n & (DK - 1);
        if (MODE == 1)
          outH[(((size_t)(b * NH + h) * SS) + s) * DK + d] = (_Float16)val;
        else
          outH[(((size_t)(b * NH + h) * DK) + d) * SS + s] = (_Float16)val;
      }
    }
  }
}

// ---------------- attention: per-wave 16 queries, loop heads+keys ------
__global__ void attention_kernel(const _Float16* __restrict__ hq,   // (B,H,S,DK)
                                 const _Float16* __restrict__ hk,   // (B,H,S,DK)
                                 const _Float16* __restrict__ hvt,  // (B,H,DK,S)
                                 const float* __restrict__ mb,      // (B,DM)
                                 const float* __restrict__ maw,     // (H)
                                 const float* __restrict__ kbias,   // (B,S) 0/-3e38
                                 _Float16* __restrict__ hatt,       // (B,S,DM)
                                 float* __restrict__ attn_out) {    // (B,S,S)
  __shared__ __attribute__((aligned(16))) _Float16 plds[8][16 * 32];
  int lane = threadIdx.x & 31;
  int wid  = threadIdx.x >> 5;
  int l16  = lane & 15;
  int hi   = lane >> 4;
  int b    = blockIdx.x / (SS / 128);
  int q0   = (blockIdx.x % (SS / 128)) * 128 + wid * 16;
  const float invscale = 0.125f;     // 1/sqrt(64)
  const float inv_h    = 1.0f / (float)NH;
  const float* kbrow = kbias + b * SS;

  for (int h = 0; h < NH; ++h) {
    const _Float16* qb = hq  + ((size_t)(b * NH + h) * SS + q0) * DK;
    const _Float16* kb = hk  + ((size_t)(b * NH + h) * SS) * DK;
    const _Float16* vb = hvt + ((size_t)(b * NH + h) * DK) * SS;
    float mawh = maw[h];

    // per-row moral bias: maw[h] * (q . mb_head), pre-scaled by 1/sqrt(dk)
    float part = 0.f;
    {
      const _Float16* qr = qb + (size_t)l16 * DK + hi * 32;
      const float* mbr = mb + (size_t)b * DM + h * DK + hi * 32;
      for (int i = 0; i < 32; ++i) part += (float)qr[i] * mbr[i];
      part += __shfl_xor(part, 16);
    }
    v8f bvs;
    for (int r = 0; r < 8; ++r)
      bvs[r] = mawh * invscale * __shfl(part, r + 8 * hi);

    // Q A-fragments (dk 0..31 and 32..63), reused for every key tile
    Frag a0, a1;
    {
      const _Float16* ap = qb + (size_t)l16 * DK + hi * 16;
      a0.u[0] = *(const uint4*)(ap);
      a0.u[1] = *(const uint4*)(ap + 8);
      a1.u[0] = *(const uint4*)(ap + 32);
      a1.u[1] = *(const uint4*)(ap + 40);
    }

    // ---- pass 1: per-lane online max/sum over this lane's columns -----
    v8f mloc, sloc;
    for (int r = 0; r < 8; ++r) { mloc[r] = -3.0e38f; sloc[r] = 0.f; }
    for (int kt = 0; kt < SS; kt += 32) {
      Frag b00, b01, b10, b11;
      const _Float16* kr0 = kb + (size_t)(kt + l16) * DK + hi * 16;
      const _Float16* kr1 = kr0 + 16 * DK;
      b00.u[0] = *(const uint4*)(kr0);
      b00.u[1] = *(const uint4*)(kr0 + 8);
      b01.u[0] = *(const uint4*)(kr0 + 32);
      b01.u[1] = *(const uint4*)(kr0 + 40);
      b10.u[0] = *(const uint4*)(kr1);
      b10.u[1] = *(const uint4*)(kr1 + 8);
      b11.u[0] = *(const uint4*)(kr1 + 32);
      b11.u[1] = *(const uint4*)(kr1 + 40);
      if (kt + 32 < SS) __builtin_prefetch(kr0 + 32 * DK, 0, 0);
      v8f acc0 = {}, acc1 = {};
      acc0 = wmma_f16(a0, b00, acc0);
      acc0 = wmma_f16(a1, b01, acc0);
      acc1 = wmma_f16(a0, b10, acc1);
      acc1 = wmma_f16(a1, b11, acc1);
      float kb0 = kbrow[kt + l16];
      float kb1 = kbrow[kt + 16 + l16];
      for (int r = 0; r < 8; ++r) {
        float t0 = fmaf(acc0[r], invscale, bvs[r] + kb0);
        float t1 = fmaf(acc1[r], invscale, bvs[r] + kb1);
        float mnew = fmaxf(mloc[r], fmaxf(t0, t1));
        float e0 = __expf(t0 - mnew);
        float e1 = __expf(t1 - mnew);
        float corr = __expf(mloc[r] - mnew);
        sloc[r] = fmaf(sloc[r], corr, e0 + e1);
        mloc[r] = mnew;
      }
    }
    // single cross-lane reduction per head (16-lane groups)
    v8f rmax, rinv;
    {
      v8f gmax = mloc;
      for (int xm = 1; xm <= 8; xm <<= 1)
        for (int r = 0; r < 8; ++r)
          gmax[r] = fmaxf(gmax[r], __shfl_xor(gmax[r], xm));
      v8f psum;
      for (int r = 0; r < 8; ++r)
        psum[r] = sloc[r] * __expf(mloc[r] - gmax[r]);
      for (int xm = 1; xm <= 8; xm <<= 1)
        for (int r = 0; r < 8; ++r)
          psum[r] += __shfl_xor(psum[r], xm);
      rmax = gmax;
      for (int r = 0; r < 8; ++r) rinv[r] = 1.0f / psum[r];
    }

    // ---- pass 2: probabilities, attn-mean accumulation, P @ V ----
    v8f o0 = {}, o1 = {}, o2 = {}, o3 = {};
    for (int kt = 0; kt < SS; kt += 32) {
      _Float16* pl = &plds[wid][0];
      for (int t2 = 0; t2 < 2; ++t2) {
        int kk = kt + t2 * 16;
        Frag bk0, bk1;
        const _Float16* kr = kb + (size_t)(kk + l16) * DK + hi * 16;
        bk0.u[0] = *(const uint4*)(kr);
        bk0.u[1] = *(const uint4*)(kr + 8);
        bk1.u[0] = *(const uint4*)(kr + 32);
        bk1.u[1] = *(const uint4*)(kr + 40);
        v8f acc = {};
        acc = wmma_f16(a0, bk0, acc);
        acc = wmma_f16(a1, bk1, acc);
        float kbv = kbrow[kk + l16];
        v8f pt;
        for (int r = 0; r < 8; ++r) {
          float tv = fmaf(acc[r], invscale, bvs[r] + kbv);
          pt[r] = __expf(tv - rmax[r]) * rinv[r];
        }
        // attn.mean over heads, exclusive region per workgroup -> deterministic
        for (int r = 0; r < 8; ++r) {
          int m = q0 + r + 8 * hi;
          size_t idx = (size_t)(b * SS + m) * SS + kk + l16;
          float v = pt[r] * inv_h;
          attn_out[idx] = (h == 0) ? v : (attn_out[idx] + v);
        }
        // stage P into LDS (C-layout -> row-major 16x32)
        for (int r = 0; r < 8; ++r)
          pl[(r + 8 * hi) * 32 + t2 * 16 + l16] = (_Float16)pt[r];
      }
      // reload P as A-fragment
      Frag ap;
      const _Float16* pr = pl + l16 * 32 + hi * 16;
      ap.u[0] = *(const uint4*)(pr);
      ap.u[1] = *(const uint4*)(pr + 8);
      // P (16x32) @ V^T tiles (32 keys x 64 dk)
      {
        Frag bf;
        const _Float16* vr;
        vr = vb + (size_t)(0  + l16) * SS + kt + hi * 16;
        bf.u[0] = *(const uint4*)(vr); bf.u[1] = *(const uint4*)(vr + 8);
        o0 = wmma_f16(ap, bf, o0);
        vr = vb + (size_t)(16 + l16) * SS + kt + hi * 16;
        bf.u[0] = *(const uint4*)(vr); bf.u[1] = *(const uint4*)(vr + 8);
        o1 = wmma_f16(ap, bf, o1);
        vr = vb + (size_t)(32 + l16) * SS + kt + hi * 16;
        bf.u[0] = *(const uint4*)(vr); bf.u[1] = *(const uint4*)(vr + 8);
        o2 = wmma_f16(ap, bf, o2);
        vr = vb + (size_t)(48 + l16) * SS + kt + hi * 16;
        bf.u[0] = *(const uint4*)(vr); bf.u[1] = *(const uint4*)(vr + 8);
        o3 = wmma_f16(ap, bf, o3);
      }
    }
    // write attended head slice (f16, (B,S,DM) row-major)
    for (int j = 0; j < 4; ++j) {
      v8f o = (j == 0) ? o0 : (j == 1) ? o1 : (j == 2) ? o2 : o3;
      int d = j * 16 + l16;
      for (int r = 0; r < 8; ++r) {
        int m = q0 + r + 8 * hi;
        hatt[(size_t)(b * SS + m) * DM + h * DK + d] = (_Float16)o[r];
      }
    }
  }
}

// ------------------------------ launch ---------------------------------
extern "C" void kernel_launch(void* const* d_in, const int* in_sizes, int n_in,
                              void* d_out, int out_size, void* d_ws, size_t ws_size,
                              hipStream_t stream) {
  const float* query = (const float*)d_in[0];
  const float* key_  = (const float*)d_in[1];
  const float* value = (const float*)d_in[2];
  const float* mc    = (const float*)d_in[3];
  const int*   mask  = (const int*)  d_in[4];
  const float* Wq    = (const float*)d_in[5];
  const float* bq    = (const float*)d_in[6];
  const float* Wk    = (const float*)d_in[7];
  const float* bk    = (const float*)d_in[8];
  const float* Wv    = (const float*)d_in[9];
  const float* bv    = (const float*)d_in[10];
  const float* Wo    = (const float*)d_in[11];
  const float* bo    = (const float*)d_in[12];
  const float* Wmc   = (const float*)d_in[13];
  const float* bmc   = (const float*)d_in[14];
  const float* maw   = (const float*)d_in[15];

  const size_t actN = (size_t)BB * SS * DM;  // 4,194,304
  const size_t wN   = (size_t)DM * DM;       // 1,048,576

  char* ws = (char*)d_ws;
  size_t off = 0;
  auto alloc = [&](size_t bytes) {
    void* p = ws + off;
    off += (bytes + 255) & ~(size_t)255;
    return p;
  };
  _Float16* qin16 = (_Float16*)alloc(actN * 2);
  _Float16* kin16 = (_Float16*)alloc(actN * 2);
  _Float16* vin16 = (_Float16*)alloc(actN * 2);
  _Float16* wq16  = (_Float16*)alloc(wN * 2);
  _Float16* wk16  = (_Float16*)alloc(wN * 2);
  _Float16* wv16  = (_Float16*)alloc(wN * 2);
  _Float16* wo16  = (_Float16*)alloc(wN * 2);
  _Float16* hq    = (_Float16*)alloc(actN * 2);
  _Float16* hk    = (_Float16*)alloc(actN * 2);
  _Float16* hvt   = (_Float16*)alloc(actN * 2);
  _Float16* hatt  = (_Float16*)alloc(actN * 2);
  float*    mbf   = (float*)   alloc((size_t)BB * DM * 4);
  float*    kbf   = (float*)   alloc((size_t)BB * SS * 4);

  float* outF     = (float*)d_out;                    // (B,S,DM)
  float* attn_out = (float*)d_out + actN;             // (B,S,S)

  auto cvt = [&](const float* s, _Float16* d, size_t n) {
    int n4 = (int)(n / 4);
    cvt_f32_f16_v4<<<(n4 + 255) / 256, 256, 0, stream>>>((const float4*)s, (v4h*)d, n4);
  };
  cvt(query, qin16, actN);
  cvt(key_,  kin16, actN);
  cvt(value, vin16, actN);
  cvt(Wq, wq16, wN);
  cvt(Wk, wk16, wN);
  cvt(Wv, wv16, wN);
  cvt(Wo, wo16, wN);

  mask_bias_kernel<<<(BB * SS + 255) / 256, 256, 0, stream>>>(mask, kbf, BB * SS);
  moral_bias_kernel<<<(BB * DM + 255) / 256, 256, 0, stream>>>(mc, Wmc, bmc, mbf);

  dim3 gg(BB * SS / 64, DM / 64);   // (64, 16)
  gemm_wmma<1><<<gg, 256, 0, stream>>>(qin16, wq16, bq, nullptr, hq,  BB * SS, DM, DM);
  gemm_wmma<1><<<gg, 256, 0, stream>>>(kin16, wk16, bk, nullptr, hk,  BB * SS, DM, DM);
  gemm_wmma<2><<<gg, 256, 0, stream>>>(vin16, wv16, bv, nullptr, hvt, BB * SS, DM, DM);

  attention_kernel<<<BB * (SS / 128), 256, 0, stream>>>(hq, hk, hvt, mbf, maw,
                                                        kbf, hatt, attn_out);

  gemm_wmma<0><<<gg, 256, 0, stream>>>(hatt, wo16, bo, outF, nullptr, BB * SS, DM, DM);
}
